// scaleDotProductAttention_18597208392387
// MI455X (gfx1250) — matrix operand order repaired and checked by reading, not compiled, not run on hardware
//
#include <hip/hip_runtime.h>
#include <hip/hip_bf16.h>

typedef __attribute__((ext_vector_type(4)))  _Float16 v4h;
typedef __attribute__((ext_vector_type(8)))  _Float16 v8h;
typedef __attribute__((ext_vector_type(16))) _Float16 v16h;
typedef __attribute__((ext_vector_type(8)))  float    v8f;
typedef __attribute__((ext_vector_type(4)))  float    v4f;
typedef __attribute__((ext_vector_type(4)))  unsigned char v4b;

#define B_   8
#define L_   4096
#define D_   64
#define SCR  4104   // padded f16 score-row stride: bank-spreads phase-3 b128 gathers

static __device__ __forceinline__ v16h cat8(v8h a, v8h b) {
  return __builtin_shufflevector(a, b, 0,1,2,3,4,5,6,7,8,9,10,11,12,13,14,15);
}
static __device__ __forceinline__ v8f wmma16(v16h a, v16h b, v8f c) {
  return __builtin_amdgcn_wmma_f32_16x16x32_f16(false, a, false, b, (short)0, c, false, false);
}

// ---- K f32 -> f16 (row major) ----------------------------------------------
__global__ __launch_bounds__(256) void cvt_k_kernel(const float* __restrict__ k,
                                                    _Float16* __restrict__ kh) {
  size_t e = ((size_t)blockIdx.x * 256 + threadIdx.x) * 4;
  v4f kv = *(const v4f*)(k + e);
  v4h h; h[0]=(_Float16)kv[0]; h[1]=(_Float16)kv[1]; h[2]=(_Float16)kv[2]; h[3]=(_Float16)kv[3];
  *(v4h*)(kh + e) = h;
}

// ---- V [B][L][D] f32 -> Vt [B][D][L] f16 (LDS tile transpose) ---------------
__global__ __launch_bounds__(256) void transpose_v_kernel(const float* __restrict__ v,
                                                          _Float16* __restrict__ vt) {
  __shared__ float tile[16][17];
  const int d0 = blockIdx.x * 16, n0 = blockIdx.y * 16, b = blockIdx.z;
  const int tx = threadIdx.x, ty = threadIdx.y;
  tile[ty][tx] = v[((size_t)b * L_ + n0 + ty) * D_ + d0 + tx];
  __syncthreads();
  vt[((size_t)b * D_ + d0 + ty) * L_ + n0 + tx] = (_Float16)tile[tx][ty];
}

// ---- fused attention: 16 query rows per workgroup ---------------------------
__global__ __launch_bounds__(256) void attn_kernel(
    const float*          __restrict__ q,
    const unsigned char*  __restrict__ mask,
    const _Float16*       __restrict__ kh,
    const _Float16*       __restrict__ vt,
    float*                __restrict__ attn_out,
    float*                __restrict__ o_out)
{
  extern __shared__ char smem[];
  _Float16* qs   = (_Float16*)smem;            // [16][72]  staged Q (padded)
  _Float16* sc   = qs + 16 * 72;               // [16][SCR] scores -> probs (f16)
  float*    obuf = (float*)(sc + 16 * SCR);    // [16][64]  O accumulator

  const int t    = threadIdx.x;
  const int lane = t & 31;
  const int wave = t >> 5;
  const int hl   = lane >> 4;                  // lane half-group (0/1)
  const int mrow = lane & 15;
  const int b    = blockIdx.x >> 8;            // 256 query tiles per batch
  const int m0   = (blockIdx.x & 255) << 4;

  for (int i = t; i < 16 * 64; i += 256) obuf[i] = 0.0f;

  { // stage Q tile f32->f16 into LDS (coalesced float4 loads)
    int e = t * 4, row = e >> 6, col = e & 63;
    v4f qv = *(const v4f*)(q + ((size_t)b * L_ + m0 + row) * D_ + col);
    v4h h; h[0]=(_Float16)qv[0]; h[1]=(_Float16)qv[1]; h[2]=(_Float16)qv[2]; h[3]=(_Float16)qv[3];
    *(v4h*)(qs + row * 72 + col) = h;
  }
  __syncthreads();

  // A operands (Q rows, per ISA 16-bit A layout), reused for all key tiles
  const _Float16* qrow = qs + mrow * 72 + hl * 8;
  const v16h aLo = cat8(*(const v8h*)(qrow),      *(const v8h*)(qrow + 16));
  const v16h aHi = cat8(*(const v8h*)(qrow + 32), *(const v8h*)(qrow + 48));

  // ---- Phase 1: S = Q*K^T, raw f16 scores into LDS --------------------------
  for (int i = 0; i < 32; ++i) {
    const int n0 = (wave * 32 + i) * 16;
    const _Float16* kp = kh + ((size_t)b * L_ + n0 + mrow) * D_ + hl * 8;
    v16h bLo = cat8(*(const v8h*)(kp),      *(const v8h*)(kp + 16));   // d 0..31
    v16h bHi = cat8(*(const v8h*)(kp + 32), *(const v8h*)(kp + 48));   // d 32..63
    v8f c = {};
    c = wmma16(aLo, bLo, c);
    c = wmma16(aHi, bHi, c);
    _Float16* srow0 = sc + (hl * 8) * SCR + n0 + mrow;
#pragma unroll
    for (int r = 0; r < 8; ++r)            // C layout: VGPR r -> M = r + 8*hl
      srow0[r * SCR] = (_Float16)c[r];
  }
  __syncthreads();

  // ---- Phase 2: mask + softmax, write attn, keep f16 probs in LDS -----------
  {
    const int rr = t >> 4, u = t & 15;
    const int gm = m0 + rr;
    const unsigned char* mp = mask + ((size_t)b * L_ + gm) * L_;
    _Float16* srow = sc + rr * SCR;
    const float NEG = -__builtin_inff();

    float mx = NEG;
    for (int j = 0; j < 64; ++j) {
      int n = (u + j * 16) * 4;
      v4b mk = __builtin_nontemporal_load((const v4b*)(mp + n));
      v4h s  = *(v4h*)(srow + n);
      float x0 = mk[0] ? NEG : (float)s[0] * 0.125f;
      float x1 = mk[1] ? NEG : (float)s[1] * 0.125f;
      float x2 = mk[2] ? NEG : (float)s[2] * 0.125f;
      float x3 = mk[3] ? NEG : (float)s[3] * 0.125f;
      v4h w; w[0]=(_Float16)x0; w[1]=(_Float16)x1; w[2]=(_Float16)x2; w[3]=(_Float16)x3;
      *(v4h*)(srow + n) = w;               // masked+scaled, in place
      mx = fmaxf(mx, fmaxf(fmaxf(x0, x1), fmaxf(x2, x3)));
    }
#pragma unroll
    for (int o = 1; o < 16; o <<= 1) mx = fmaxf(mx, __shfl_xor(mx, o, 32));

    float sum = 0.0f;
    for (int j = 0; j < 64; ++j) {
      int n = (u + j * 16) * 4;
      v4h s = *(v4h*)(srow + n);
      sum += __expf((float)s[0] - mx) + __expf((float)s[1] - mx)
           + __expf((float)s[2] - mx) + __expf((float)s[3] - mx);
    }
#pragma unroll
    for (int o = 1; o < 16; o <<= 1) sum += __shfl_xor(sum, o, 32);
    const float inv = 1.0f / sum;

    float* arow = attn_out + ((size_t)b * L_ + gm) * L_;
    for (int j = 0; j < 64; ++j) {
      int n = (u + j * 16) * 4;
      v4h s = *(v4h*)(srow + n);
      float p0 = __expf((float)s[0] - mx) * inv;
      float p1 = __expf((float)s[1] - mx) * inv;
      float p2 = __expf((float)s[2] - mx) * inv;
      float p3 = __expf((float)s[3] - mx) * inv;
      v4f pv; pv[0]=p0; pv[1]=p1; pv[2]=p2; pv[3]=p3;
      __builtin_nontemporal_store(pv, (v4f*)(arow + n));   // streaming attn write
      v4h w; w[0]=(_Float16)p0; w[1]=(_Float16)p1; w[2]=(_Float16)p2; w[3]=(_Float16)p3;
      *(v4h*)(srow + n) = w;                               // probs for phase 3
    }
  }
  __syncthreads();

  // ---- Phase 3: O = P * V (4 d-group accumulators per wave) -----------------
  v8f c0 = {}, c1 = {}, c2 = {}, c3 = {};
  for (int i = 0; i < 16; ++i) {
    const int n0 = (wave * 16 + i) * 32;
    const _Float16* pr = sc + mrow * SCR + n0 + hl * 8;
    v16h aP = cat8(*(const v8h*)(pr), *(const v8h*)(pr + 16));
    const _Float16* vb = vt + ((size_t)(b * D_ + mrow)) * L_ + n0 + hl * 8;
    v16h b0 = cat8(*(const v8h*)(vb),               *(const v8h*)(vb + 16));
    v16h b1 = cat8(*(const v8h*)(vb + 16 * L_),     *(const v8h*)(vb + 16 * L_ + 16));
    v16h b2 = cat8(*(const v8h*)(vb + 32 * L_),     *(const v8h*)(vb + 32 * L_ + 16));
    v16h b3 = cat8(*(const v8h*)(vb + 48 * L_),     *(const v8h*)(vb + 48 * L_ + 16));
    c0 = wmma16(aP, b0, c0);
    c1 = wmma16(aP, b1, c1);
    c2 = wmma16(aP, b2, c2);
    c3 = wmma16(aP, b3, c3);
  }
  {
    float* ob = obuf + (hl * 8) * 64 + mrow;
#pragma unroll
    for (int r = 0; r < 8; ++r) {          // ds_add_f32 cross-wave reduction
      atomicAdd(ob + r * 64 +  0, c0[r]);
      atomicAdd(ob + r * 64 + 16, c1[r]);
      atomicAdd(ob + r * 64 + 32, c2[r]);
      atomicAdd(ob + r * 64 + 48, c3[r]);
    }
  }
  __syncthreads();
  {
    int e = t * 4;
    v4f o = *(const v4f*)(obuf + e);
    *(v4f*)(o_out + ((size_t)b * L_ + m0) * D_ + e) = o;
  }
}

extern "C" void kernel_launch(void* const* d_in, const int* in_sizes, int n_in,
                              void* d_out, int out_size, void* d_ws, size_t ws_size,
                              hipStream_t stream) {
  (void)in_sizes; (void)n_in; (void)out_size; (void)ws_size;
  const float*         q    = (const float*)d_in[0];
  const float*         k    = (const float*)d_in[1];
  const float*         v    = (const float*)d_in[2];
  const unsigned char* mask = (const unsigned char*)d_in[3];   // jax bool = 1 byte

  float* attn_out = (float*)d_out;
  float* o_out    = attn_out + (size_t)B_ * L_ * L_;

  _Float16* kh  = (_Float16*)d_ws;                    // 4 MB
  _Float16* vtr = kh + (size_t)B_ * L_ * D_;          // 4 MB

  cvt_k_kernel<<<(B_ * L_ * D_) / 1024, 256, 0, stream>>>(k, kh);
  transpose_v_kernel<<<dim3(D_ / 16, L_ / 16, B_), dim3(16, 16), 0, stream>>>(v, vtr);

  const size_t smem = (size_t)(16 * 72 + 16 * SCR) * sizeof(_Float16)
                    + (size_t)16 * 64 * sizeof(float);          // ~137.7 KB
  hipFuncSetAttribute((const void*)attn_kernel,
                      hipFuncAttributeMaxDynamicSharedMemorySize, (int)smem);
  attn_kernel<<<B_ * (L_ / 16), 256, smem, stream>>>(q, mask, kh, vtr, attn_out, o_out);
}
